// GaussianImage_34789235097895
// MI455X (gfx1250) — compile-verified, matrix operand order checked by reading, not run on hardware
//
#include <hip/hip_runtime.h>
#include <math.h>

typedef __attribute__((ext_vector_type(16))) _Float16 v16h;
typedef __attribute__((ext_vector_type(8)))  float    v8f;

#define ALPHA_LO (1.0f / 255.0f)
#define ALPHA_HI 0.999f
#define CH 128  // gaussians staged per barrier (4 WMMA sub-chunks of 32)

// ---------------------------------------------------------------------------
// Pass 1: per-gaussian preprocessing (tiny: N~1e4). Produces SoA arrays in ws:
//   ws[0..4] = cx, cy, conic a, conic b, conic c
//   ws[5..7] = r, g, b colors (feats * sigmoid(e8))
// Padding rows [N, NP) get zero colors -> contribute nothing to the image.
// ---------------------------------------------------------------------------
__global__ __launch_bounds__(256) void gi_preprocess(
    const float* __restrict__ embed, float* __restrict__ ws,
    int N, int NP, int H, int W) {
  int i = blockIdx.x * blockDim.x + threadIdx.x;
  if (i >= NP) return;
  float cx = 0.f, cy = 0.f, ca = 0.f, cb = 0.f, cc = 0.f;
  float r = 0.f, g = 0.f, bl = 0.f;
  if (i < N) {
    const float* e = embed + (size_t)i * 9;
    float mx = tanhf(e[0]);
    float my = tanhf(e[1]);
    cx = 0.5f * (mx + 1.0f) * (float)W;
    cy = 0.5f * (my + 1.0f) * (float)H;
    float l1 = e[5] + 0.5f, l2 = e[6], l3 = e[7] + 0.5f;
    float c00 = l1 * l1;
    float c01 = l1 * l2;
    float c11 = l2 * l2 + l3 * l3;
    float det = c00 * c11 - c01 * c01;
    float inv = 1.0f / det;
    ca = c11 * inv;
    cb = -c01 * inv;
    cc = c00 * inv;
    float u = 1.0f / (1.0f + __expf(-e[8]));
    r  = e[2] * u;
    g  = e[3] * u;
    bl = e[4] * u;
  }
  ws[0 * NP + i] = cx;
  ws[1 * NP + i] = cy;
  ws[2 * NP + i] = ca;
  ws[3 * NP + i] = cb;
  ws[4 * NP + i] = cc;
  ws[5 * NP + i] = r;
  ws[6 * NP + i] = g;
  ws[7 * NP + i] = bl;
}

// ---------------------------------------------------------------------------
// Pass 2: splat. GEMM view: img(M=pix, N=3ch) += alpha(M, K=N_gauss) x color(K, 3).
// Block = 256 threads = 8 wave32s; each wave owns a 16-pixel row segment
// (M=16), block covers a 16x8 tile. K consumed in 128-gaussian staged chunks
// (one barrier pair each), 4x v_wmma_f32_16x16x32_f16 per chunk with an fp32
// accumulator. Alphas (v_exp_f32, the throughput limiter) stay in fp32 VALU;
// the rank update rides the matrix pipe which co-executes with VALU.
//
// A-fragment lane layout (ISA 7.12.2, 16-bit A 16x32): lane L -> M = L&15,
// half h = L>>4; VGPR j holds K pair: j<4 -> K = 2j + 8h, j>=4 -> K = 16 +
// 2(j-4) + 8h. B mirrors with lane -> N (color channel; cols 3..15 read a
// zero row via a branch-free LDS pointer select). C/D: lane -> N = L&15,
// VGPR r -> M = r + 8*(L>>4).
// ---------------------------------------------------------------------------
__global__ __launch_bounds__(256) void gi_render(
    const float* __restrict__ ws, float* __restrict__ out,
    int NP, int H, int W) {
  __shared__ float sg[8][CH];  // staged chunk: cx,cy,a,b,c,r,g,b
  __shared__ float zrow[CH];   // zeros for B columns 3..15

  const int lane = threadIdx.x & 31;
  const int wave = threadIdx.x >> 5;
  const int m    = lane & 15;   // A: pixel in segment; B/C: column N
  const int h    = lane >> 4;   // half-wave: K subset / M-row group

  const int x0 = blockIdx.x * 16;
  const int y  = blockIdx.y * 8 + wave;
  const float px = (float)(x0 + m);
  const float py = (float)y;

  // branch-free color source: channels 0..2 -> color rows, 3..15 -> zero row
  const float* colp = (m < 3) ? &sg[5 + m][0] : &zrow[0];

  // cooperative fill mapping: thread -> one float4 of the 4KB chunk
  const int t4  = threadIdx.x * 4;  // 0..1020
  const int arr = t4 >> 7;          // /CH -> SoA array 0..7
  const int idx = t4 & (CH - 1);    // offset within chunk (multiple of 4)

  if (threadIdx.x < CH) zrow[threadIdx.x] = 0.0f;  // one-time init

  v8f acc = {};  // 16x16 f32 accumulator (rows = pixels, cols 0..2 = RGB)

  for (int kc = 0; kc < NP; kc += CH) {
    // stage 128 gaussians (4KB) into LDS: global_load_b128 + ds_store_b128
    const float* src = ws + (size_t)arr * NP + kc + idx;
    *reinterpret_cast<float4*>(&sg[arr][idx]) =
        *reinterpret_cast<const float4*>(src);
    __syncthreads();

    // speculative prefetch of next chunk (silently dropped past the end)
    __builtin_prefetch(src + CH, 0, 1);

#pragma unroll
    for (int kb = 0; kb < CH; kb += 32) {
      v16h afrag = {};
      v16h bfrag = {};
#pragma unroll
      for (int j = 0; j < 8; ++j) {
        const int kk = kb + ((j < 4) ? (2 * j) : (16 + 2 * (j - 4))) + 8 * h;
#pragma unroll
        for (int t = 0; t < 2; ++t) {
          const int k = kk + t;
          // ---- A: alpha(pixel m, gaussian kc+k) ----
          float dx = sg[0][k] - px;
          float dy = sg[1][k] - py;
          float sa = sg[2][k];
          float sb = sg[3][k];
          float sc = sg[4][k];
          float sigma = 0.5f * sa * dx * dx + sb * dx * dy + 0.5f * sc * dy * dy;
          float al = __expf(-sigma);             // v_exp_f32
          al = fminf(al, ALPHA_HI);
          al = (sigma >= 0.0f && al >= ALPHA_LO) ? al : 0.0f;
          afrag[2 * j + t] = (_Float16)al;
          // ---- B: color[channel = m][gaussian kc+k] (branch-free) ----
          bfrag[2 * j + t] = (_Float16)colp[k];
        }
      }
      // rank-32 update on the matrix pipe: acc += A(16x32) x B(32x16)
      acc = __builtin_amdgcn_wmma_f32_16x16x32_f16(
          /*neg_a=*/false, afrag, /*neg_b=*/false, bfrag,
          /*c_mod=*/(short)0, acc, /*reuse_a=*/false, /*reuse_b=*/false);
    }

    __syncthreads();  // protect sg before next chunk overwrites it
  }

  // write-out: C/D layout -> lane holds column N=m, rows M = h*8 + r
  if (m < 3 && y < H) {
#pragma unroll
    for (int r = 0; r < 8; ++r) {
      int x = x0 + h * 8 + r;
      out[((size_t)m * H + y) * W + x] = acc[r];
    }
  }
}

// ---------------------------------------------------------------------------
// Host launcher. H/W derived from out_size (3*H*W, square image) since device
// scalar inputs cannot be read during graph capture.
// ---------------------------------------------------------------------------
extern "C" void kernel_launch(void* const* d_in, const int* in_sizes, int n_in,
                              void* d_out, int out_size, void* d_ws, size_t ws_size,
                              hipStream_t stream) {
  const float* embed = (const float*)d_in[0];
  const int N  = in_sizes[0] / 9;
  const int NP = (N + CH - 1) & ~(CH - 1);  // pad K to staging chunk size

  const int HW  = out_size / 3;
  const int dim = (int)(sqrt((double)HW) + 0.5);  // 512 for the reference
  const int H = dim, W = dim;

  float* ws = (float*)d_ws;  // needs 8 * NP * 4 bytes (~324 KB)

  gi_preprocess<<<(NP + 255) / 256, 256, 0, stream>>>(embed, ws, N, NP, H, W);

  dim3 grid(W / 16, H / 8);   // 32 x 64 = 2048 blocks of 8 waves
  gi_render<<<grid, 256, 0, stream>>>(ws, (float*)d_out, NP, H, W);
}
